// ami_att_net2_82832739270711
// MI455X (gfx1250) — matrix operand order repaired and checked by reading, not compiled, not run on hardware
//
#include <hip/hip_runtime.h>

// ---------------------------------------------------------------------------
// GNN forward (drug-protein) for MI455X / gfx1250.
// Dense layers: bf16 WMMA (v_wmma_f32_16x16x32_bf16), f32 accumulate.
//   - 64x128 block tile, 8 waves (2x4), 32x32 per wave => 4 WMMA / K-step
//   - uniform fast path: b128 global loads, b128 LDS stores/loads
//   - B tile stored transposed in LDS so fragments are contiguous b128 loads
// Graph ops: f32 atomics; edge softmax via CAS float-max + exp-sum.
// ---------------------------------------------------------------------------

#define B_  256
#define N1_ 10240
#define E1_ 81920
#define N3_ 25600
#define E3_ 819200
#define D_  128

typedef __attribute__((ext_vector_type(4)))  float          v4f;
typedef __attribute__((ext_vector_type(8)))  float          v8f;
typedef __attribute__((ext_vector_type(16))) __bf16         v16bf;
typedef __attribute__((ext_vector_type(16))) unsigned short v16u;
typedef __attribute__((ext_vector_type(8)))  unsigned short v8us;

__device__ __forceinline__ unsigned short f2bf(float f) {
  unsigned int u = __float_as_uint(f);
  u += 0x7FFFu + ((u >> 16) & 1u);           // round-to-nearest-even
  return (unsigned short)(u >> 16);
}

__device__ __forceinline__ void atomicMaxF(float* addr, float val) {
  unsigned int* ai = (unsigned int*)addr;
  unsigned int old = __hip_atomic_load(ai, __ATOMIC_RELAXED, __HIP_MEMORY_SCOPE_AGENT);
  while (__uint_as_float(old) < val) {
    unsigned int assumed = old;
    old = atomicCAS(ai, assumed, __float_as_uint(val));
    if (old == assumed) break;
  }
}

// Build a 16-lane bf16 fragment from two contiguous 8x bf16 runs in LDS.
__device__ __forceinline__ v16bf frag16(const unsigned short* p0,
                                        const unsigned short* p1) {
  v8us lo = *(const v8us*)p0;      // ds_load_b128
  v8us hi = *(const v8us*)p1;      // ds_load_b128
  v16u r;
  #pragma unroll
  for (int j = 0; j < 8; ++j) { r[j] = lo[j]; r[8 + j] = hi[j]; }
  return __builtin_bit_cast(v16bf, r);
}

// ---------------------------------------------------------------------------
// Tiled bf16 WMMA GEMM:  C[M,N](ldc) = act( A[M,K] @ B[K,N] + bias )
// ---------------------------------------------------------------------------
#define BM 64
#define BN 128
#define BK 32
#define LDA 40   // ushorts per A row (32 + 8 pad, keeps 16B alignment)
#define LDB 40   // ushorts per Bs row (Bs is transposed: [n][k])

__global__ __launch_bounds__(256) void gemm_bf16_wmma(
    const float* __restrict__ A, const float* __restrict__ Bm,
    const float* __restrict__ bias, float* __restrict__ C,
    int M, int K, int N, int ldc, int relu)
{
  __shared__ unsigned short As[BM * LDA];    // [m][k]
  __shared__ unsigned short Bs[BN * LDB];    // [n][k] (transposed)

  const int tid  = threadIdx.x;
  const int wave = tid >> 5;
  const int lane = tid & 31;
  const int lh   = lane >> 4;                // lane half (0/1)
  const int q    = lane & 15;
  const int wmo  = (wave & 1) * 32;          // wave M offset inside tile
  const int wno  = (wave >> 1) * 32;         // wave N offset inside tile
  const int m0   = blockIdx.y * BM;
  const int n0   = blockIdx.x * BN;

  // staging coordinates
  const int ar = tid >> 2;                   // A row 0..63
  const int ac = (tid & 3) * 8;              // A col 0,8,16,24
  const int bk = tid >> 3;                   // B k-row 0..31
  const int bc = (tid & 7) * 16;             // B col 0..112

  const bool fast_mn = (m0 + BM <= M) && (n0 + BN <= N) &&
                       ((K & 3) == 0) && ((N & 3) == 0);

  v8f acc00 = {}, acc01 = {}, acc10 = {}, acc11 = {};

  for (int k0 = 0; k0 < K; k0 += BK) {
    if (fast_mn && (k0 + BK <= K)) {
      // ---------- fast path: vector loads, packed LDS stores ----------
      const float* pa = A + (long)(m0 + ar) * K + k0 + ac;
      v4f fa0 = *(const v4f*)pa;
      v4f fa1 = *(const v4f*)(pa + 4);
      v8us pk;
      #pragma unroll
      for (int j = 0; j < 4; ++j) { pk[j] = f2bf(fa0[j]); pk[4 + j] = f2bf(fa1[j]); }
      *(v8us*)&As[ar * LDA + ac] = pk;       // ds_store_b128

      const float* pb = Bm + (long)(k0 + bk) * N + n0 + bc;
      v4f fb0 = *(const v4f*)pb;
      v4f fb1 = *(const v4f*)(pb + 4);
      v4f fb2 = *(const v4f*)(pb + 8);
      v4f fb3 = *(const v4f*)(pb + 12);
      float bv[16];
      #pragma unroll
      for (int j = 0; j < 4; ++j) {
        bv[j] = fb0[j]; bv[4 + j] = fb1[j]; bv[8 + j] = fb2[j]; bv[12 + j] = fb3[j];
      }
      #pragma unroll
      for (int u = 0; u < 16; ++u)
        Bs[(bc + u) * LDB + bk] = f2bf(bv[u]);   // transposed store

      if (k0 + 2 * BK <= K) {                 // prefetch next K tile
        __builtin_prefetch(pa + BK, 0, 1);
        __builtin_prefetch(pb + (long)BK * N, 0, 1);
      }
    } else {
      // ---------- slow path: clamped addresses + select (branch-free) ----------
      {
        const int gr  = m0 + ar;
        const int grc = gr < M ? gr : M - 1;
        #pragma unroll
        for (int u = 0; u < 8; ++u) {
          const int gk  = k0 + ac + u;
          const int gkc = gk < K ? gk : K - 1;
          float v = A[(long)grc * K + gkc];
          if (gr >= M || gk >= K) v = 0.f;
          As[ar * LDA + ac + u] = f2bf(v);
        }
      }
      {
        const int gk  = k0 + bk;
        const int gkc = gk < K ? gk : K - 1;
        #pragma unroll
        for (int u = 0; u < 16; ++u) {
          const int gn  = n0 + bc + u;
          const int gnc = gn < N ? gn : N - 1;
          float v = Bm[(long)gkc * N + gnc];
          if (gk >= K || gn >= N) v = 0.f;
          Bs[(bc + u) * LDB + bk] = f2bf(v);
        }
      }
    }
    __syncthreads();

    // ---- fragments (contiguous b128 LDS loads per ISA 7.12.2 layouts) ----
    const int ma0 = (wmo + q) * LDA;
    const int ma1 = (wmo + 16 + q) * LDA;
    v16bf a0 = frag16(&As[ma0 + lh * 8], &As[ma0 + 16 + lh * 8]);
    v16bf a1 = frag16(&As[ma1 + lh * 8], &As[ma1 + 16 + lh * 8]);
    const int nb0 = (wno + q) * LDB;
    const int nb1 = (wno + 16 + q) * LDB;
    v16bf b0 = frag16(&Bs[nb0 + lh * 8], &Bs[nb0 + 16 + lh * 8]);
    v16bf b1 = frag16(&Bs[nb1 + lh * 8], &Bs[nb1 + 16 + lh * 8]);

    acc00 = __builtin_amdgcn_wmma_f32_16x16x32_bf16(false, a0, false, b0, (short)0, acc00, false, false);
    acc01 = __builtin_amdgcn_wmma_f32_16x16x32_bf16(false, a0, false, b1, (short)0, acc01, false, false);
    acc10 = __builtin_amdgcn_wmma_f32_16x16x32_bf16(false, a1, false, b0, (short)0, acc10, false, false);
    acc11 = __builtin_amdgcn_wmma_f32_16x16x32_bf16(false, a1, false, b1, (short)0, acc11, false, false);
    __syncthreads();
  }

  // ---- store: VGPR r -> row (r + lh*8); col = lane&15 ----
  const int colA = n0 + wno + q;
  const int colB = n0 + wno + 16 + q;
  const int rowA = m0 + wmo + lh * 8;
  const int rowB = m0 + wmo + 16 + lh * 8;
  const bool full_store = (m0 + BM <= M) && (n0 + BN <= N);

  #pragma unroll
  for (int r = 0; r < 8; ++r) {
    float v00 = acc00[r], v01 = acc01[r], v10 = acc10[r], v11 = acc11[r];
    if (bias) {
      const float ba = (colA < N) ? bias[colA] : 0.f;
      const float bb = (colB < N) ? bias[colB] : 0.f;
      v00 += ba; v01 += bb; v10 += ba; v11 += bb;
    }
    if (relu) {
      v00 = fmaxf(v00, 0.f); v01 = fmaxf(v01, 0.f);
      v10 = fmaxf(v10, 0.f); v11 = fmaxf(v11, 0.f);
    }
    if (full_store) {
      C[(long)(rowA + r) * ldc + colA] = v00;
      C[(long)(rowA + r) * ldc + colB] = v01;
      C[(long)(rowB + r) * ldc + colA] = v10;
      C[(long)(rowB + r) * ldc + colB] = v11;
    } else {
      if (rowA + r < M && colA < N) C[(long)(rowA + r) * ldc + colA] = v00;
      if (rowA + r < M && colB < N) C[(long)(rowA + r) * ldc + colB] = v01;
      if (rowB + r < M && colA < N) C[(long)(rowB + r) * ldc + colA] = v10;
      if (rowB + r < M && colB < N) C[(long)(rowB + r) * ldc + colB] = v11;
    }
  }
}

// ---------------------------------------------------------------------------
// Utility / graph kernels
// ---------------------------------------------------------------------------
__global__ void fill_f32(float* p, long n, float v) {
  long i = (long)blockIdx.x * blockDim.x + threadIdx.x;
  if (i < n) p[i] = v;
}

__global__ void set_col128(float* dst, const float* src, int n) {
  int i = blockIdx.x * blockDim.x + threadIdx.x;
  if (i < n) dst[(long)i * 128 + 127] = src[i];
}

__global__ void deg_accum(const int* col, const float* w, float* deg, int E) {
  int e = blockIdx.x * blockDim.x + threadIdx.x;
  if (e < E) atomicAdd(&deg[col[e]], w[e]);
}

__global__ void rsqrt_inplace(float* d, int n) {
  int i = blockIdx.x * blockDim.x + threadIdx.x;
  if (i < n) {
    float v = d[i];
    d[i] = (v > 0.f) ? rsqrtf(fmaxf(v, 1e-12f)) : 0.f;
  }
}

// out[col] += h[row] * dinv[row]*w*dinv[col]; edges e>=E are self loops (w=1)
__global__ void gcn_scatter(const float* __restrict__ h, float* __restrict__ out,
                            const int* __restrict__ row, const int* __restrict__ col,
                            const float* __restrict__ w, const float* __restrict__ dinv,
                            int E, int NT) {
  long gid = (long)blockIdx.x * blockDim.x + threadIdx.x;
  long total = (long)(E + NT) * 128;
  if (gid >= total) return;
  int e = (int)(gid >> 7);
  int f = (int)(gid & 127);
  int r, c; float ww;
  if (e < E) { r = row[e]; c = col[e]; ww = w[e]; }
  else       { r = c = e - E; ww = 1.f; }
  float norm = dinv[r] * ww * dinv[c];
  atomicAdd(&out[(long)c * 128 + f], h[(long)r * 128 + f] * norm);
}

__global__ void bias_relu(float* x, const float* bias, long rows, int F) {
  long i = (long)blockIdx.x * blockDim.x + threadIdx.x;
  if (i < rows * F) {
    int f = (int)(i % F);
    x[i] = fmaxf(x[i] + bias[f], 0.f);
  }
}

__global__ void pool_accum(const float* x, const int* batch, float* pool, long n, int F) {
  long i = (long)blockIdx.x * blockDim.x + threadIdx.x;
  if (i < n * F) {
    long node = i / F; int f = (int)(i % F);
    atomicAdd(&pool[(long)batch[node] * F + f], x[i]);
  }
}

__global__ void count_accum(const int* batch, float* cnt, long n) {
  long i = (long)blockIdx.x * blockDim.x + threadIdx.x;
  if (i < n) atomicAdd(&cnt[batch[i]], 1.f);
}

__global__ void pool_div(const float* pool, const float* cnt, float* outp, int nb, int F) {
  int i = blockIdx.x * blockDim.x + threadIdx.x;
  if (i < nb * F) outp[i] = pool[i] / fmaxf(cnt[i / F], 1.f);
}

__global__ void axpy_rowvec(float* x, const float* vec, float s, long rows, int F) {
  long i = (long)blockIdx.x * blockDim.x + threadIdx.x;
  if (i < rows * F) x[i] += s * vec[i % F];
}

__global__ void node_dots(const float* h, const float* as, const float* ad,
                          float* ss, float* sd, int n, int F) {
  int i = blockIdx.x * blockDim.x + threadIdx.x;
  if (i >= n) return;
  float a = 0.f, b = 0.f;
  const float* hp = h + (long)i * F;
  for (int f = 0; f < F; ++f) { a += hp[f] * as[f]; b += hp[f] * ad[f]; }
  ss[i] = a; sd[i] = b;
}

__global__ void edge_max(const int* src, const int* dst, const float* ss,
                         const float* sd, float* mx, int E, int NT) {
  int e = blockIdx.x * blockDim.x + threadIdx.x;
  if (e >= E + NT) return;
  int s, d;
  if (e < E) { s = src[e]; d = dst[e]; } else { s = d = e - E; }
  float l = ss[s] + sd[d];
  l = (l > 0.f) ? l : 0.2f * l;              // leaky_relu 0.2
  atomicMaxF(&mx[d], l);
}

__global__ void edge_exp(const int* src, const int* dst, const float* ss,
                         const float* sd, const float* mx, float* ebuf,
                         float* ssum, int E, int NT) {
  int e = blockIdx.x * blockDim.x + threadIdx.x;
  if (e >= E + NT) return;
  int s, d;
  if (e < E) { s = src[e]; d = dst[e]; } else { s = d = e - E; }
  float l = ss[s] + sd[d];
  l = (l > 0.f) ? l : 0.2f * l;
  float ex = __expf(l - mx[d]);
  ebuf[e] = ex;
  atomicAdd(&ssum[d], ex);
}

__global__ void gat_scatter(const float* __restrict__ h, const int* __restrict__ src,
                            const int* __restrict__ dst, const float* __restrict__ ebuf,
                            const float* __restrict__ ssum, float* __restrict__ out,
                            int E, int NT) {
  long gid = (long)blockIdx.x * blockDim.x + threadIdx.x;
  long total = (long)(E + NT) * 128;
  if (gid >= total) return;
  int e = (int)(gid >> 7);
  int f = (int)(gid & 127);
  int s, d;
  if (e < E) { s = src[e]; d = dst[e]; } else { s = d = e - E; }
  float alpha = ebuf[e] / ssum[d];
  atomicAdd(&out[(long)d * 128 + f], h[(long)s * 128 + f] * alpha);
}

__global__ void bn_relu(float* x, const float* g, const float* b,
                        const float* m, const float* v, long rows, int F) {
  long i = (long)blockIdx.x * blockDim.x + threadIdx.x;
  if (i < rows * F) {
    int f = (int)(i % F);
    float y = (x[i] - m[f]) * rsqrtf(v[f] + 1e-5f) * g[f] + b[f];
    x[i] = fmaxf(y, 0.f);
  }
}

__global__ void relu_copy(const float* x, float* y, long n) {
  long i = (long)blockIdx.x * blockDim.x + threadIdx.x;
  if (i < n) y[i] = fmaxf(x[i], 0.f);
}

__global__ void add_gather(const float* x11, const float* drug, const int* batch,
                           float* out, long rows, int F) {
  long i = (long)blockIdx.x * blockDim.x + threadIdx.x;
  if (i < rows * F) {
    long node = i / F; int f = (int)(i % F);
    out[i] = x11[i] + 0.2f * drug[(long)batch[node] * F + f];
  }
}

__global__ void concat2(const float* a, const float* b, float* z, int nb, int F) {
  int i = blockIdx.x * blockDim.x + threadIdx.x;
  if (i < nb * 2 * F) {
    int rrow = i / (2 * F), c = i % (2 * F);
    z[i] = (c < F) ? a[(long)rrow * F + c] : b[(long)rrow * F + (c - F)];
  }
}

__global__ void final_out(const float* z, const float* w, const float* b,
                          float* out, int nb, int F) {
  int i = blockIdx.x * blockDim.x + threadIdx.x;
  if (i >= nb) return;
  float acc = 0.f;
  const float* zp = z + (long)i * F;
  for (int f = 0; f < F; ++f) acc += zp[f] * w[f];
  out[i] = acc + b[0];
}

// ---------------------------------------------------------------------------
// Host orchestration
// ---------------------------------------------------------------------------
static inline unsigned cdiv(long a, int b) { return (unsigned)((a + b - 1) / b); }

static void gemm(const float* A, const float* Bm, const float* bias, float* C,
                 int M, int K, int N, int ldc, int relu, hipStream_t s) {
  dim3 grid(cdiv(N, BN), cdiv(M, BM));
  gemm_bf16_wmma<<<grid, 256, 0, s>>>(A, Bm, bias, C, M, K, N, ldc, relu);
}

extern "C" void kernel_launch(void* const* d_in, const int* in_sizes, int n_in,
                              void* d_out, int out_size, void* d_ws, size_t ws_size,
                              hipStream_t stream) {
  (void)in_sizes; (void)n_in; (void)out_size; (void)ws_size;
  const float* x1      = (const float*)d_in[0];
  const float* x3      = (const float*)d_in[1];
  const float* ami_dis = (const float*)d_in[2];
  const float* ami_li  = (const float*)d_in[3];
  const float* fc00_w  = (const float*)d_in[4];  const float* fc00_b = (const float*)d_in[5];
  const float* fc01_w  = (const float*)d_in[6];  const float* fc01_b = (const float*)d_in[7];
  const float* fc02_w  = (const float*)d_in[8];  const float* fc02_b = (const float*)d_in[9];
  const float* w1      = (const float*)d_in[10]; const float* b1     = (const float*)d_in[11];
  const float* w2      = (const float*)d_in[12]; const float* b2     = (const float*)d_in[13];
  const float* fc03_w  = (const float*)d_in[14]; const float* fc03_b = (const float*)d_in[15];
  const float* gat_W   = (const float*)d_in[16];
  const float* gat_as  = (const float*)d_in[17];
  const float* gat_ad  = (const float*)d_in[18];
  const float* gat_bias= (const float*)d_in[19];
  const float* vn_emb  = (const float*)d_in[20];
  const float* vl1_w   = (const float*)d_in[21]; const float* vl1_b  = (const float*)d_in[22];
  const float* bn1_g   = (const float*)d_in[23]; const float* bn1_b  = (const float*)d_in[24];
  const float* bn1_m   = (const float*)d_in[25]; const float* bn1_v  = (const float*)d_in[26];
  const float* vl2_w   = (const float*)d_in[27]; const float* vl2_b  = (const float*)d_in[28];
  const float* bn2_g   = (const float*)d_in[29]; const float* bn2_b  = (const float*)d_in[30];
  const float* bn2_m   = (const float*)d_in[31]; const float* bn2_v  = (const float*)d_in[32];
  const float* gcn_W   = (const float*)d_in[33]; const float* gcn_b  = (const float*)d_in[34];
  const float* fc1_w   = (const float*)d_in[35]; const float* fc1_b  = (const float*)d_in[36];
  const float* fc2_w   = (const float*)d_in[37]; const float* fc2_b  = (const float*)d_in[38];
  const float* fc3_w   = (const float*)d_in[39]; const float* fc3_b  = (const float*)d_in[40];
  const float* out_w   = (const float*)d_in[41]; const float* out_b  = (const float*)d_in[42];
  const int*   drug_ei = (const int*)d_in[43];   // [2][E1]
  const int*   ami_ei  = (const int*)d_in[44];   // [2][E3]
  const int*   batch1  = (const int*)d_in[45];
  const int*   ami_bat = (const int*)d_in[46];

  float* ws   = (float*)d_ws;
  float* outp = (float*)d_out;          // [0..255] out, [256..] h2 (256x512)

  const int* dr_src = drug_ei;           const int* dr_dst = drug_ei + E1_;
  const int* am_row = ami_ei;            const int* am_col = ami_ei + E3_;

  // ---- workspace layout (floats) ----
  const size_t A0   = 0;                      // 25600*1024
  const size_t A1   = 26214400;               // 25600*512
  // region [0, 39321600) reused after protein MLP:
  const size_t HB   = 0;                      // 25600*128 (gemm h)
  const size_t OB   = 3276800;                // gcn out ping
  const size_t OB2  = 6553600;                // gcn out pong
  const size_t DEG  = 9830400;                // 25600 (deg -> dinv)
  const size_t PLA  = 9856000;                // 256*128 protein pool acc
  const size_t CNT  = 9888768;                // 256
  const size_t H0   = 9889024;                // 10240*128
  const size_t H1   = 11199744;               // 10240*64
  const size_t XD   = 11855104;               // 10240*128 (drug x)
  const size_t HG   = 13165824;               // 10240*128 (gat h)
  const size_t SS   = 14476544;               // 10240
  const size_t SD   = 14486784;
  const size_t MX   = 14497024;
  const size_t SSM  = 14507264;
  const size_t EB   = 14517504;               // 92160
  const size_t GO   = 14609664;               // 10240*128 (gat out 2/3)
  const size_t X11  = 15920384;               // 10240*128
  const size_t XGA  = 17231104;               // 256*128
  const size_t CND  = 17263872;               // 256
  const size_t TSM  = 17264128;               // 256*128
  const size_t TB1  = 17296896;               // 256*256
  const size_t TB2  = 17362432;               // 256*128
  const size_t AMI  = 17395200;               // 256*128
  const size_t XG   = 17427968;               // 256*128
  const size_t Z    = 17460736;               // 256*256
  const size_t Z2   = 17526272;               // 256*512
  const size_t Z3   = 17657344;               // 256*256
  const size_t Z4   = 17722880;               // 256*128
  const size_t AF   = 39321600;               // 25600*128 protein features

  const int TE1 = E1_ + N1_;                  // drug edges incl self loops
  const long NF3 = (long)N3_ * 128;
  const long NF1 = (long)N1_ * 128;

  // ================= protein branch =================
  gemm(x3, fc00_w, fc00_b, ws + A0, N3_, 1900, 1024, 1024, 1, stream);
  gemm(ws + A0, fc01_w, fc01_b, ws + A1, N3_, 1024, 512, 512, 1, stream);
  gemm(ws + A1, fc02_w, fc02_b, ws + AF, N3_, 512, 127, 128, 1, stream);
  set_col128<<<cdiv(N3_, 256), 256, 0, stream>>>(ws + AF, ami_li, N3_);

  // degree norm (self loops weight 1 -> init 1.0)
  fill_f32<<<cdiv(N3_, 256), 256, 0, stream>>>(ws + DEG, N3_, 1.f);
  deg_accum<<<cdiv(E3_, 256), 256, 0, stream>>>(am_col, ami_dis, ws + DEG, E3_);
  rsqrt_inplace<<<cdiv(N3_, 256), 256, 0, stream>>>(ws + DEG, N3_);

  fill_f32<<<cdiv(B_ * 128, 256), 256, 0, stream>>>(ws + PLA, B_ * 128, 0.f);
  fill_f32<<<1, 256, 0, stream>>>(ws + CNT, B_, 0.f);
  count_accum<<<cdiv(N3_, 256), 256, 0, stream>>>(ami_bat, ws + CNT, N3_);

  const size_t gcn_in[3]  = { AF, OB, OB2 };
  const size_t gcn_out[3] = { OB, OB2, OB };
  for (int l = 0; l < 3; ++l) {
    gemm(ws + gcn_in[l], gcn_W + (size_t)l * 128 * 128, nullptr, ws + HB,
         N3_, 128, 128, 128, 0, stream);
    fill_f32<<<cdiv(NF3, 256), 256, 0, stream>>>(ws + gcn_out[l], NF3, 0.f);
    gcn_scatter<<<cdiv((long)(E3_ + N3_) * 128, 256), 256, 0, stream>>>(
        ws + HB, ws + gcn_out[l], am_row, am_col, ami_dis, ws + DEG, E3_, N3_);
    bias_relu<<<cdiv(NF3, 256), 256, 0, stream>>>(ws + gcn_out[l], gcn_b + l * 128, N3_, 128);
    pool_accum<<<cdiv(NF3, 256), 256, 0, stream>>>(ws + gcn_out[l], ami_bat, ws + PLA, N3_, 128);
  }
  pool_div<<<cdiv(B_ * 128, 256), 256, 0, stream>>>(ws + PLA, ws + CNT, ws + AMI, B_, 128);

  // ================= drug branch =================
  gemm(x1, w1, b1, ws + H0, N1_, 52, 128, 128, 1, stream);
  gemm(ws + H0, w2, b2, ws + H1, N1_, 128, 64, 64, 1, stream);
  gemm(ws + H1, fc03_w, fc03_b, ws + XD, N1_, 64, 128, 128, 0, stream);
  axpy_rowvec<<<cdiv(NF1, 256), 256, 0, stream>>>(ws + XD, vn_emb, 0.2f, N1_, 128);

  // GAT layer runner
  auto run_gat = [&](int l, size_t xin, size_t outb) {
    gemm(ws + xin, gat_W + (size_t)l * 128 * 128, nullptr, ws + HG, N1_, 128, 128, 128, 0, stream);
    node_dots<<<cdiv(N1_, 256), 256, 0, stream>>>(ws + HG, gat_as + l * 128,
                                                  gat_ad + l * 128, ws + SS, ws + SD, N1_, 128);
    fill_f32<<<cdiv(N1_, 256), 256, 0, stream>>>(ws + MX, N1_, -1e30f);
    fill_f32<<<cdiv(N1_, 256), 256, 0, stream>>>(ws + SSM, N1_, 0.f);
    edge_max<<<cdiv(TE1, 256), 256, 0, stream>>>(dr_src, dr_dst, ws + SS, ws + SD, ws + MX, E1_, N1_);
    edge_exp<<<cdiv(TE1, 256), 256, 0, stream>>>(dr_src, dr_dst, ws + SS, ws + SD,
                                                 ws + MX, ws + EB, ws + SSM, E1_, N1_);
    fill_f32<<<cdiv(NF1, 256), 256, 0, stream>>>(ws + outb, NF1, 0.f);
    gat_scatter<<<cdiv((long)TE1 * 128, 256), 256, 0, stream>>>(
        ws + HG, dr_src, dr_dst, ws + EB, ws + SSM, ws + outb, E1_, N1_);
    bias_relu<<<cdiv(NF1, 256), 256, 0, stream>>>(ws + outb, gat_bias + l * 128, N1_, 128);
  };

  // layer 1 -> X11
  run_gat(0, XD, X11);

  fill_f32<<<cdiv(B_ * 128, 256), 256, 0, stream>>>(ws + XGA, B_ * 128, 0.f);
  fill_f32<<<1, 256, 0, stream>>>(ws + CND, B_, 0.f);
  count_accum<<<cdiv(N1_, 256), 256, 0, stream>>>(batch1, ws + CND, N1_);
  pool_accum<<<cdiv(NF1, 256), 256, 0, stream>>>(ws + X11, batch1, ws + XGA, N1_, 128);

  // virtual-node update: t = segsum(x11) + vn_emb -> MLP+BN -> drug vec
  fill_f32<<<cdiv(B_ * 128, 256), 256, 0, stream>>>(ws + TSM, B_ * 128, 0.f);
  pool_accum<<<cdiv(NF1, 256), 256, 0, stream>>>(ws + X11, batch1, ws + TSM, N1_, 128);
  axpy_rowvec<<<cdiv(B_ * 128, 256), 256, 0, stream>>>(ws + TSM, vn_emb, 1.f, B_, 128);
  gemm(ws + TSM, vl1_w, vl1_b, ws + TB1, B_, 128, 256, 256, 0, stream);
  bn_relu<<<cdiv(B_ * 256, 256), 256, 0, stream>>>(ws + TB1, bn1_g, bn1_b, bn1_m, bn1_v, B_, 256);
  gemm(ws + TB1, vl2_w, vl2_b, ws + TB2, B_, 256, 128, 128, 0, stream);
  bn_relu<<<cdiv(B_ * 128, 256), 256, 0, stream>>>(ws + TB2, bn2_g, bn2_b, bn2_m, bn2_v, B_, 128);

  // x = x11 + 0.2*drug[batch1]  (overwrite XD)
  add_gather<<<cdiv(NF1, 256), 256, 0, stream>>>(ws + X11, ws + TB2, batch1, ws + XD, N1_, 128);

  // layers 2 and 3 (both consume the same updated x)
  run_gat(1, XD, GO);
  pool_accum<<<cdiv(NF1, 256), 256, 0, stream>>>(ws + GO, batch1, ws + XGA, N1_, 128);
  run_gat(2, XD, GO);
  pool_accum<<<cdiv(NF1, 256), 256, 0, stream>>>(ws + GO, batch1, ws + XGA, N1_, 128);

  pool_div<<<cdiv(B_ * 128, 256), 256, 0, stream>>>(ws + XGA, ws + CND, ws + XG, B_, 128);

  // ================= fused head =================
  concat2<<<cdiv(B_ * 256, 256), 256, 0, stream>>>(ws + XG, ws + AMI, ws + Z, B_, 128);
  // h2 = z @ fc1 + b  -> second output (pre-relu), written directly to d_out
  gemm(ws + Z, fc1_w, fc1_b, outp + B_, B_, 256, 512, 512, 0, stream);
  relu_copy<<<cdiv((long)B_ * 512, 256), 256, 0, stream>>>(outp + B_, ws + Z2, (long)B_ * 512);
  gemm(ws + Z2, fc2_w, fc2_b, ws + Z3, B_, 512, 256, 256, 1, stream);
  gemm(ws + Z3, fc3_w, fc3_b, ws + Z4, B_, 256, 128, 128, 1, stream);
  final_out<<<1, 256, 0, stream>>>(ws + Z4, out_w, out_b, outp, B_, 128);
}
